// DiscreteSender_16209206575124
// MI455X (gfx1250) — compile-verified, matrix operand order checked by reading, not run on hardware
//
#include <hip/hip_runtime.h>
#include <math.h>

// ---------------------------------------------------------------------------
// DiscreteSender on MI455X (gfx1250, wave32).
// GEMMs: V_WMMA_F32_16X16X32_BF16 with double-buffered LDS staging fed by
// GLOBAL_LOAD_ASYNC_TO_LDS_B128 (ASYNCcnt) -- block-level A/B reuse cuts L2
// traffic 8x vs direct-global fragments, making the decode compute-bound.
// Sequential 16-step LSTM decode -> chain of kernels on `stream`.
// ---------------------------------------------------------------------------

typedef __attribute__((ext_vector_type(16))) __bf16 v16bf;
typedef __attribute__((ext_vector_type(8)))  __bf16 v8bf;
typedef __attribute__((ext_vector_type(8)))  float  v8f;

#define BB   2048
#define AA   8
#define EIN  64
#define EOUT 256
#define HH   1024
#define VV   1024
#define LL   16

static __device__ __forceinline__ __bf16 f2bf(float f) {
    unsigned u = __builtin_bit_cast(unsigned, f);
    unsigned r = u + 0x7FFFu + ((u >> 16) & 1u);   // round-to-nearest-even
    unsigned short s = (unsigned short)(r >> 16);
    return __builtin_bit_cast(__bf16, s);
}

// 16B global -> LDS async copy (per lane). CDNA5 async path, ASYNCcnt-tracked.
static __device__ __forceinline__ void async_copy16(unsigned lds_off,
                                                    const __bf16* gptr) {
    asm volatile("global_load_async_to_lds_b128 %0, %1, off"
                 :: "v"(lds_off), "v"(gptr) : "memory");
}

// -------------------------- conversion / init -----------------------------

__global__ void cvt_f32_bf16_kernel(const float* __restrict__ src,
                                    __bf16* __restrict__ dst, int n) {
    int i = blockIdx.x * blockDim.x + threadIdx.x;
    if (i < n) dst[i] = f2bf(src[i]);
}

__global__ void zero_f32_kernel(float* __restrict__ p, int n) {
    int i = blockIdx.x * blockDim.x + threadIdx.x;
    if (i < n) p[i] = 0.0f;
}

__global__ void sos_init_kernel(const float* __restrict__ sos,
                                __bf16* __restrict__ emb, int n) {
    int i = blockIdx.x * blockDim.x + threadIdx.x;
    if (i < n) emb[i] = f2bf(sos[i & (EOUT - 1)]);
}

__global__ void embed_gather_kernel(const int* __restrict__ x,
                                    const float* __restrict__ in_emb,
                                    __bf16* __restrict__ ie, int total) {
    int idx = blockIdx.x * blockDim.x + threadIdx.x;
    if (idx >= total) return;
    int e  = idx & (EIN - 1);
    int ba = idx >> 6;
    int tok = x[ba];
    ie[idx] = f2bf(in_emb[tok * EIN + e]);
}

// ------------------------------ WMMA GEMM ----------------------------------
// C[M,N](f32, row stride ldc) = A[M,K](bf16) * W[N,K](bf16)^T + bias[N] (+C)
// Block: 256 threads = 8 waves stacked along M; block tile 128(M) x 64(N).
// K staged in 64-wide tiles through double-buffered LDS via async DMA:
//   sA: 128x64 bf16 (16KB)  sB: 64x64 bf16 (8KB)  x2 buffers = 48KB LDS.
// Each wave: 16x64 strip, 4 f32 accumulators, 8 WMMA per K-tile.
__global__ __launch_bounds__(256) void gemm_bf16_wmma_kernel(
    const __bf16* __restrict__ A, const __bf16* __restrict__ W,
    const float* __restrict__ bias, float* __restrict__ C,
    int M, int N, int K, int ldc, int accumulate)
{
    __shared__ __bf16 sA[2][128 * 64];
    __shared__ __bf16 sB[2][64 * 64];

    const int tid  = threadIdx.x;
    const int lane = tid & 31;
    const int wave = tid >> 5;
    const int lm   = lane & 15;      // row (A) / col (B,C) within 16-tile
    const int half = lane >> 4;      // K-half selector (ISA bf16 frag layout)
    const int Mbase = blockIdx.x * 128;
    const int Nbase = blockIdx.y * 64;
    const int KT = K >> 6;           // number of 64-wide K tiles

    // stage K-tile kt into buffer buf: 6 async b128 issues per wave
    auto issue_tile = [&](int buf, int kt) {
        unsigned aB = (unsigned)(size_t)(&sA[buf][0]);
        unsigned bB = (unsigned)(size_t)(&sB[buf][0]);
        const int k64 = kt * 64;
#pragma unroll
        for (int i = 0; i < 4; ++i) {            // A: 1024 chunks of 16B
            int chunk = i * 256 + tid;
            int row = chunk >> 3, c = chunk & 7;
            async_copy16(aB + chunk * 16,
                         A + (size_t)(Mbase + row) * K + k64 + c * 8);
        }
#pragma unroll
        for (int i = 0; i < 2; ++i) {            // B: 512 chunks of 16B
            int chunk = i * 256 + tid;
            int row = chunk >> 3, c = chunk & 7;
            async_copy16(bB + chunk * 16,
                         W + (size_t)(Nbase + row) * K + k64 + c * 8);
        }
    };

    v8f acc[4] = {{}, {}, {}, {}};

    issue_tile(0, 0);
    for (int kt = 0; kt < KT; ++kt) {
        const int cur = kt & 1;
        if (kt + 1 < KT) {
            issue_tile(cur ^ 1, kt + 1);
            asm volatile("s_wait_asynccnt 0x6" ::: "memory");  // cur tile done
        } else {
            asm volatile("s_wait_asynccnt 0x0" ::: "memory");
        }
        __syncthreads();    // all waves' async writes for cur tile visible

        const __bf16* sa = &sA[cur][0];
        const __bf16* sb = &sB[cur][0];
#pragma unroll
        for (int ks = 0; ks < 64; ks += 32) {
            // A fragment: lanes 0-15 K={0..7,16..23}, lanes 16-31 K={8..15,24..31}
            const int arow = (wave * 16 + lm) * 64 + ks + half * 8;
            v8bf alo = *(const v8bf*)(sa + arow);
            v8bf ahi = *(const v8bf*)(sa + arow + 16);
            v16bf a;
#pragma unroll
            for (int i = 0; i < 8; ++i) { a[i] = alo[i]; a[8 + i] = ahi[i]; }
#pragma unroll
            for (int nt = 0; nt < 4; ++nt) {
                const int brow = (nt * 16 + lm) * 64 + ks + half * 8;
                v8bf blo = *(const v8bf*)(sb + brow);
                v8bf bhi = *(const v8bf*)(sb + brow + 16);
                v16bf b;
#pragma unroll
                for (int i = 0; i < 8; ++i) { b[i] = blo[i]; b[8 + i] = bhi[i]; }
                acc[nt] = __builtin_amdgcn_wmma_f32_16x16x32_bf16(
                    false, a, false, b, (short)0, acc[nt], false, false);
            }
        }
        __syncthreads();    // all reads of cur done before it is overwritten
    }

    // epilogue: C layout -> lane holds column, VGPR v -> row half*8+v
#pragma unroll
    for (int nt = 0; nt < 4; ++nt) {
        int col = Nbase + nt * 16 + lm;
        float bv = bias ? bias[col] : 0.0f;
#pragma unroll
        for (int v = 0; v < 8; ++v) {
            int r = Mbase + wave * 16 + half * 8 + v;
            float* cp = C + (size_t)r * ldc + col;
            float val = acc[nt][v] + bv;
            if (accumulate) val += *cp;
            *cp = val;
        }
    }
}

// ------------------------------ LSTM cell ----------------------------------

__global__ void lstm_cell_kernel(const float* __restrict__ gates,
                                 float* __restrict__ c,
                                 __bf16* __restrict__ h_bf) {
    int idx = blockIdx.x * blockDim.x + threadIdx.x;
    if (idx >= BB * HH) return;
    int b = idx >> 10;
    int j = idx & (HH - 1);
    const float* g = gates + (size_t)b * 4 * HH;
    float ig = g[j], fg = g[HH + j], gg = g[2 * HH + j], og = g[3 * HH + j];
    float si = 1.0f / (1.0f + __expf(-ig));
    float sf = 1.0f / (1.0f + __expf(-fg));
    float so = 1.0f / (1.0f + __expf(-og));
    float cn = sf * c[idx] + si * tanhf(gg);
    float hn = so * tanhf(cn);
    c[idx]   = cn;
    h_bf[idx] = f2bf(hn);
}

// -------------------------- argmax + feedback ------------------------------

__global__ __launch_bounds__(256) void argmax_emb_kernel(
    const float* __restrict__ logits_t, int ldrow,
    float* __restrict__ seq_t,
    const __bf16* __restrict__ out_emb_bf,
    __bf16* __restrict__ emb)
{
    const int lane = threadIdx.x & 31;
    const int wave = threadIdx.x >> 5;
    const int b    = blockIdx.x * 8 + wave;
    if (b >= BB) return;

    const float* row = logits_t + (size_t)b * ldrow;
    float bv = -INFINITY;
    int   bi = 0;
    for (int j = lane; j < VV; j += 32) {
        float v = row[j];
        if (v > bv) { bv = v; bi = j; }
    }
#pragma unroll
    for (int off = 16; off > 0; off >>= 1) {
        float v2 = __shfl_xor(bv, off, 32);
        int   i2 = __shfl_xor(bi, off, 32);
        if (v2 > bv || (v2 == bv && i2 < bi)) { bv = v2; bi = i2; }
    }
    if (lane == 0) seq_t[(size_t)b * LL] = (float)bi;

    const __bf16* er = out_emb_bf + (size_t)bi * EOUT;
    __bf16*       eo = emb + (size_t)b * EOUT;
    for (int j = lane; j < EOUT; j += 32) eo[j] = er[j];
}

// ------------------------------ launcher -----------------------------------

extern "C" void kernel_launch(void* const* d_in, const int* in_sizes, int n_in,
                              void* d_out, int out_size, void* d_ws, size_t ws_size,
                              hipStream_t stream) {
    (void)in_sizes; (void)n_in; (void)out_size; (void)ws_size;
    const int*   x      = (const int*)  d_in[0];
    const float* in_emb = (const float*)d_in[1];
    const float* outEmb = (const float*)d_in[2];
    const float* W_in   = (const float*)d_in[3];
    const float* b_in   = (const float*)d_in[4];
    const float* W_ih   = (const float*)d_in[5];
    const float* b_ih   = (const float*)d_in[6];
    const float* W_hh   = (const float*)d_in[7];
    const float* b_hh   = (const float*)d_in[8];
    const float* W_out  = (const float*)d_in[9];
    const float* b_out  = (const float*)d_in[10];
    const float* sos    = (const float*)d_in[11];

    float* out      = (float*)d_out;            // [B*L] seq ++ [B*L*V] logits
    float* logits0  = out + (size_t)BB * LL;

    char* ws = (char*)d_ws;
    size_t off = 0;
    auto alloc = [&](size_t bytes) { void* p = ws + off; off = (off + bytes + 255) & ~(size_t)255; return p; };
    __bf16* wIn_bf   = (__bf16*)alloc((size_t)HH * AA * EIN * 2);
    __bf16* wIh_bf   = (__bf16*)alloc((size_t)4 * HH * EOUT * 2);
    __bf16* wHh_bf   = (__bf16*)alloc((size_t)4 * HH * HH * 2);
    __bf16* wOut_bf  = (__bf16*)alloc((size_t)VV * HH * 2);
    __bf16* oEmb_bf  = (__bf16*)alloc((size_t)VV * EOUT * 2);
    __bf16* ie_bf    = (__bf16*)alloc((size_t)BB * AA * EIN * 2);
    __bf16* h_bf     = (__bf16*)alloc((size_t)BB * HH * 2);
    float*  c_f32    = (float*) alloc((size_t)BB * HH * 4);
    __bf16* emb_bf   = (__bf16*)alloc((size_t)BB * EOUT * 2);
    float*  h0_f32   = (float*) alloc((size_t)BB * HH * 4);
    float*  gates    = (float*) alloc((size_t)BB * 4 * HH * 4);

    const int T = 256;
    auto cvt = [&](const float* s, __bf16* d, int n) {
        cvt_f32_bf16_kernel<<<(n + T - 1) / T, T, 0, stream>>>(s, d, n);
    };

    cvt(W_in,  wIn_bf,  HH * AA * EIN);
    cvt(W_ih,  wIh_bf,  4 * HH * EOUT);
    cvt(W_hh,  wHh_bf,  4 * HH * HH);
    cvt(W_out, wOut_bf, VV * HH);
    cvt(outEmb, oEmb_bf, VV * EOUT);

    embed_gather_kernel<<<(BB * AA * EIN + T - 1) / T, T, 0, stream>>>(
        x, in_emb, ie_bf, BB * AA * EIN);

    // h0 = ie @ W_in^T + b_in   (M=2048, N=1024, K=512)
    gemm_bf16_wmma_kernel<<<dim3(BB / 128, HH / 64), 256, 0, stream>>>(
        ie_bf, wIn_bf, b_in, h0_f32, BB, HH, AA * EIN, HH, 0);
    cvt(h0_f32, h_bf, BB * HH);

    zero_f32_kernel<<<(BB * HH + T - 1) / T, T, 0, stream>>>(c_f32, BB * HH);
    sos_init_kernel<<<(BB * EOUT + T - 1) / T, T, 0, stream>>>(sos, emb_bf, BB * EOUT);

    for (int t = 0; t < LL; ++t) {
        gemm_bf16_wmma_kernel<<<dim3(BB / 128, (4 * HH) / 64), 256, 0, stream>>>(
            emb_bf, wIh_bf, b_ih, gates, BB, 4 * HH, EOUT, 4 * HH, 0);
        gemm_bf16_wmma_kernel<<<dim3(BB / 128, (4 * HH) / 64), 256, 0, stream>>>(
            h_bf, wHh_bf, b_hh, gates, BB, 4 * HH, HH, 4 * HH, 1);
        lstm_cell_kernel<<<(BB * HH + T - 1) / T, T, 0, stream>>>(gates, c_f32, h_bf);
        gemm_bf16_wmma_kernel<<<dim3(BB / 128, VV / 64), 256, 0, stream>>>(
            h_bf, wOut_bf, b_out, logits0 + (size_t)t * VV, BB, VV, HH, LL * VV, 0);
        argmax_emb_kernel<<<BB / 8, 256, 0, stream>>>(
            logits0 + (size_t)t * VV, LL * VV, out + t, oEmb_bf, emb_bf);
    }
}
